// Dilated_res_block_4217657885151
// MI455X (gfx1250) — matrix operand order repaired and checked by reading, not compiled, not run on hardware
//
#include <hip/hip_runtime.h>
#include <hip/hip_bf16.h>
#include <math.h>

typedef __attribute__((ext_vector_type(16))) _Float16 v16h;
typedef __attribute__((ext_vector_type(8)))  float    v8f;
typedef __attribute__((ext_vector_type(4)))  int      v4i;
typedef __attribute__((ext_vector_type(4)))  unsigned int u32x4;

#define BB     2
#define NN     20480
#define KK_NB  16
#define DIN    64
#define DOUT   128
#define MP     (BB*NN)          /* 40960 points           */
#define MNK    (BB*NN*KK_NB)    /* 655360 (point,neighbor)*/
#define EPSBN  1e-5f

#if __has_builtin(__builtin_amdgcn_global_load_async_to_lds_b128)
#define HAVE_ASYNC_LDS 1
#endif

// 16-byte global -> LDS copy; async (ASYNCcnt-tracked) on gfx1250.
__device__ __forceinline__ void cp16_g2l(_Float16* ldst, const _Float16* gsrc) {
#ifdef HAVE_ASYNC_LDS
    __builtin_amdgcn_global_load_async_to_lds_b128(
        (__attribute__((address_space(1))) v4i*)gsrc,
        (__attribute__((address_space(3))) v4i*)ldst, 0, 0);
#else
    *(u32x4*)ldst = *(const u32x4*)gsrc;
#endif
}

__device__ __forceinline__ void async_copies_done() {
#ifdef HAVE_ASYNC_LDS
#if __has_builtin(__builtin_amdgcn_s_wait_asynccnt)
    __builtin_amdgcn_s_wait_asynccnt(0);
#else
    asm volatile("s_wait_asynccnt 0x0" ::: "memory");
#endif
#endif
}

// ---------------------------------------------------------------- utilities

__global__ void zero_stats_kernel(float* __restrict__ s) {
    int i = blockIdx.x * 256 + threadIdx.x;
    if (i < 4096) s[i] = 0.f;
}

// relative position encoding -> rp16[MNK][32] f16 (10 used, rest zero)
__global__ void relpos_kernel(const float* __restrict__ xyz,
                              const long long* __restrict__ idx,
                              _Float16* __restrict__ rp) {
    int m = blockIdx.x * 256 + threadIdx.x;
    if (m >= MNK) return;
    int pn = m / KK_NB;                 // b*NN + n
    int b  = pn / NN;
    long long j = idx[m];
    const float* pc = xyz + (size_t)pn * 3;
    const float* nc = xyz + ((size_t)b * NN + (size_t)j) * 3;
    float rx = pc[0] - nc[0], ry = pc[1] - nc[1], rz = pc[2] - nc[2];
    float d  = sqrtf(rx*rx + ry*ry + rz*rz);
    _Float16* o = rp + (size_t)m * 32;
    o[0]=(_Float16)d;  o[1]=(_Float16)rx; o[2]=(_Float16)ry; o[3]=(_Float16)rz;
    o[4]=(_Float16)pc[0]; o[5]=(_Float16)pc[1]; o[6]=(_Float16)pc[2];
    o[7]=(_Float16)nc[0]; o[8]=(_Float16)nc[1]; o[9]=(_Float16)nc[2];
    #pragma unroll
    for (int c = 10; c < 32; ++c) o[c] = (_Float16)0.f;
}

// ------------------------------------------------------- WMMA GEMM + BN stat
// Y[M, coutTotal] (chunk [coutOff, coutOff+NTILE)) = A[M, CINP] * W^T
// CHMAJOR=0: A16 row-major f16, stride CINP (pre-padded) -> async LDS copy
// CHMAJOR=1: Af32 channel-major [B, CIN_REAL, Npts]      -> scalar staging
// accumulates per-channel sum / sum^2 into stats[0..2*coutTotal)
template<int CINP, int NTILE, int CHMAJOR, int CIN_REAL>
__global__ void __launch_bounds__(256)
gemm_bn_kernel(const _Float16* __restrict__ A16, const float* __restrict__ Af32,
               int Npts, const float* __restrict__ W,
               float* __restrict__ Y, float* __restrict__ stats,
               int coutTotal, int M) {
    constexpr int NSUB = NTILE / 16;
    __shared__ __align__(16) _Float16 Wl[NTILE][CINP];
    __shared__ __align__(16) _Float16 Al[128][32];

    const int tid  = threadIdx.x;
    const int lane = tid & 31, wave = tid >> 5;
    const int half = lane >> 4, lrow = lane & 15;
    const int m0      = blockIdx.x * 128;
    const int coutOff = blockIdx.y * NTILE;

    for (int i = tid; i < NTILE * CINP; i += 256) {
        int o = i / CINP, c = i % CINP;
        float w = (c < CIN_REAL) ? W[(size_t)(coutOff + o) * CIN_REAL + c] : 0.f;
        Wl[o][c] = (_Float16)w;
    }
    __syncthreads();

    v8f acc[NSUB];
    #pragma unroll
    for (int t = 0; t < NSUB; ++t)
        #pragma unroll
        for (int e = 0; e < 8; ++e) acc[t][e] = 0.f;

    #pragma unroll
    for (int kk = 0; kk < CINP / 32; ++kk) {
        if (CHMAJOR) {
            for (int i = tid; i < 128 * 32; i += 256) {
                int r = i >> 5, c = i & 31;
                int gc = kk * 32 + c;
                int m  = m0 + r;
                float v = 0.f;
                if (gc < CIN_REAL) {
                    int b = m / Npts, n = m - b * Npts;
                    v = Af32[((size_t)b * CIN_REAL + gc) * Npts + n];
                }
                Al[r][c] = (_Float16)v;
            }
        } else {
            // 128 rows x 32 f16 = 128 x 4 chunks of 16B, contiguous per row
            for (int i = tid; i < 128 * 4; i += 256) {
                int r = i >> 2, ch = i & 3;
                const _Float16* g =
                    A16 + (size_t)(m0 + r) * CINP + kk * 32 + ch * 8;
                cp16_g2l(&Al[r][ch * 8], g);
            }
            async_copies_done();
        }
        __syncthreads();

        v16h a;
        #pragma unroll
        for (int v = 0; v < 8; ++v) {     // ISA 16-bit A 16x32 layout
            int kl = ((v >> 2) << 4) + (half << 3) + ((v & 3) << 1);
            a[2*v]   = Al[wave * 16 + lrow][kl];
            a[2*v+1] = Al[wave * 16 + lrow][kl + 1];
        }
        #pragma unroll
        for (int nt = 0; nt < NSUB; ++nt) {
            v16h bv;
            #pragma unroll
            for (int v = 0; v < 8; ++v) { // ISA 16-bit B 32x16 layout
                int kl = ((v >> 2) << 4) + (half << 3) + ((v & 3) << 1);
                bv[2*v]   = Wl[nt * 16 + lrow][kk * 32 + kl];
                bv[2*v+1] = Wl[nt * 16 + lrow][kk * 32 + kl + 1];
            }
            acc[nt] = __builtin_amdgcn_wmma_f32_16x16x32_f16(
                false, a, false, bv, (short)0, acc[nt], false, false);
        }
        __syncthreads();
    }

    #pragma unroll
    for (int nt = 0; nt < NSUB; ++nt) {
        int col = coutOff + nt * 16 + lrow;
        float s = 0.f, s2 = 0.f;
        #pragma unroll
        for (int v = 0; v < 8; ++v) {     // C/D layout: M = v + 8*half
            int m = m0 + wave * 16 + v + half * 8;
            float y = acc[nt][v];
            Y[(size_t)m * coutTotal + col] = y;
            s += y; s2 += y * y;
        }
        s  += __shfl_xor(s, 16, 32);
        s2 += __shfl_xor(s2, 16, 32);
        if (half == 0) {
            atomicAdd(&stats[col], s);
            atomicAdd(&stats[coutTotal + col], s2);
        }
    }
}

// -------------------------------- fused attentive pooling (concat + scores +
// softmax over K + weighted sum), one wave per point (16 neighbor rows)
__global__ void __launch_bounds__(256)
att_pool_kernel(const _Float16* __restrict__ gsrc,  // [MP, 64] point features
                const _Float16* __restrict__ xsrc,  // [MNK, 64] geo features
                const long long* __restrict__ idx,
                const float* __restrict__ Wfc,      // [128,128]
                _Float16* __restrict__ agg) {       // [MP, 128]
    __shared__ __align__(16) _Float16 Wl[128][128];
    __shared__ __align__(16) _Float16 Al[128][32];

    const int tid  = threadIdx.x;
    const int lane = tid & 31, wave = tid >> 5;
    const int half = lane >> 4, lrow = lane & 15;
    const int g0 = blockIdx.x * 8;                // first point of block

    for (int i = tid; i < 128 * 128; i += 256)
        Wl[i >> 7][i & 127] = (_Float16)Wfc[i];
    __syncthreads();

    v8f acc[8];
    #pragma unroll
    for (int t = 0; t < 8; ++t)
        #pragma unroll
        for (int e = 0; e < 8; ++e) acc[t][e] = 0.f;

    #pragma unroll
    for (int kk = 0; kk < 4; ++kk) {
        // stage A rows of the virtual concat [gather(gsrc) | xsrc]; each half
        // of a row is 128B contiguous -> 16B async copies
        for (int i = tid; i < 128 * 4; i += 256) {
            int r = i >> 2, ch = i & 3;
            int pt = g0 + (r >> 4);
            int k  = r & 15;
            size_t mnk = (size_t)pt * KK_NB + k;
            int gc = kk * 32 + ch * 8;
            const _Float16* g;
            if (gc < 64) {
                int b = pt / NN;
                long long j = idx[mnk];
                g = gsrc + ((size_t)b * NN + (size_t)j) * 64 + gc;
            } else {
                g = xsrc + mnk * 64 + (gc - 64);
            }
            cp16_g2l(&Al[r][ch * 8], g);
        }
        async_copies_done();
        __syncthreads();

        v16h a;
        #pragma unroll
        for (int v = 0; v < 8; ++v) {
            int kl = ((v >> 2) << 4) + (half << 3) + ((v & 3) << 1);
            a[2*v]   = Al[wave * 16 + lrow][kl];
            a[2*v+1] = Al[wave * 16 + lrow][kl + 1];
        }
        #pragma unroll
        for (int nt = 0; nt < 8; ++nt) {
            v16h bv;
            #pragma unroll
            for (int v = 0; v < 8; ++v) {
                int kl = ((v >> 2) << 4) + (half << 3) + ((v & 3) << 1);
                bv[2*v]   = Wl[nt * 16 + lrow][kk * 32 + kl];
                bv[2*v+1] = Wl[nt * 16 + lrow][kk * 32 + kl + 1];
            }
            acc[nt] = __builtin_amdgcn_wmma_f32_16x16x32_f16(
                false, a, false, bv, (short)0, acc[nt], false, false);
        }
        __syncthreads();
    }

    // softmax over the 16 neighbors of this wave's point, then weighted sum.
    // lanes l and l^16 hold rows k=v and k=v+8 of the same column.
    const int pt = g0 + wave;
    const int b  = pt / NN;
    #pragma unroll
    for (int nt = 0; nt < 8; ++nt) {
        int col = nt * 16 + lrow;
        float mx = -1e30f;
        #pragma unroll
        for (int v = 0; v < 8; ++v) mx = fmaxf(mx, acc[nt][v]);
        mx = fmaxf(mx, __shfl_xor(mx, 16, 32));
        float e[8], s = 0.f;
        #pragma unroll
        for (int v = 0; v < 8; ++v) { e[v] = __expf(acc[nt][v] - mx); s += e[v]; }
        s += __shfl_xor(s, 16, 32);
        float inv = 1.f / s;
        float p = 0.f;
        #pragma unroll
        for (int v = 0; v < 8; ++v) {
            int k = v + half * 8;
            size_t mnk = (size_t)pt * KK_NB + k;
            float feat;
            if (col < 64) {
                long long j = idx[mnk];
                feat = (float)gsrc[((size_t)b * NN + (size_t)j) * 64 + col];
            } else {
                feat = (float)xsrc[mnk * 64 + (col - 64)];
            }
            p += e[v] * inv * feat;
        }
        p += __shfl_xor(p, 16, 32);
        if (half == 0) agg[(size_t)pt * 128 + col] = (_Float16)p;
    }
}

// ---------------------------------------------- BN finalize: f32 lin -> f16
__global__ void bn_finalize_kernel(const float* __restrict__ Y,
                                   _Float16* __restrict__ O,
                                   const float* __restrict__ stats,
                                   const float* __restrict__ gamma,
                                   const float* __restrict__ beta,
                                   int COUT, int M, int relu) {
    int i = blockIdx.x * 256 + threadIdx.x;
    if (i >= M * COUT) return;
    int c = i % COUT;
    float cnt  = (float)M;
    float mean = stats[c] / cnt;
    float var  = stats[COUT + c] / cnt - mean * mean;
    float y = (Y[i] - mean) * rsqrtf(var + EPSBN) * gamma[c] + beta[c];
    if (relu) y = fmaxf(y, 0.f);
    O[i] = (_Float16)y;
}

// ------------------------------------- BN(mlp2) + BN(shortcut) + leaky ReLU
__global__ void final_kernel(const float* __restrict__ y1,
                             const float* __restrict__ y2,
                             const float* __restrict__ st1,
                             const float* __restrict__ st2,
                             const float* __restrict__ g1, const float* __restrict__ b1,
                             const float* __restrict__ g2, const float* __restrict__ b2,
                             float* __restrict__ out) {
    int i = blockIdx.x * 256 + threadIdx.x;
    if (i >= MP * 256) return;
    int c = i & 255, m = i >> 8;
    float cnt = (float)MP;
    float m1 = st1[c] / cnt, v1 = st1[256 + c] / cnt - m1 * m1;
    float a  = (y1[i] - m1) * rsqrtf(v1 + EPSBN) * g1[c] + b1[c];
    float m2 = st2[c] / cnt, v2 = st2[256 + c] / cnt - m2 * m2;
    float d  = (y2[i] - m2) * rsqrtf(v2 + EPSBN) * g2[c] + b2[c];
    float s = a + d;
    s = (s > 0.f) ? s : 0.2f * s;
    int bb = m / NN, n = m - bb * NN;
    out[((size_t)bb * 256 + c) * NN + n] = s;  // [B, 2*DOUT, N, 1]
}

// --------------------------------------------------------------------- host
extern "C" void kernel_launch(void* const* d_in, const int* in_sizes, int n_in,
                              void* d_out, int out_size, void* d_ws, size_t ws_size,
                              hipStream_t stream) {
    const float*     feature = (const float*)d_in[0];
    const float*     xyz     = (const float*)d_in[1];
    const long long* neigh   = (const long long*)d_in[2];
    const float* W_mlp1 = (const float*)d_in[3];
    const float* g_mlp1 = (const float*)d_in[4];  const float* b_mlp1 = (const float*)d_in[5];
    const float* W_bb1  = (const float*)d_in[6];
    const float* g_bb1  = (const float*)d_in[7];  const float* b_bb1  = (const float*)d_in[8];
    const float* W_a1fc = (const float*)d_in[9];
    const float* W_a1m  = (const float*)d_in[10];
    const float* g_a1   = (const float*)d_in[11]; const float* b_a1   = (const float*)d_in[12];
    const float* W_bb2  = (const float*)d_in[13];
    const float* g_bb2  = (const float*)d_in[14]; const float* b_bb2  = (const float*)d_in[15];
    const float* W_a2fc = (const float*)d_in[16];
    const float* W_a2m  = (const float*)d_in[17];
    const float* g_a2   = (const float*)d_in[18]; const float* b_a2   = (const float*)d_in[19];
    const float* W_mlp2 = (const float*)d_in[20];
    const float* g_mlp2 = (const float*)d_in[21]; const float* b_mlp2 = (const float*)d_in[22];
    const float* W_sc   = (const float*)d_in[23];
    const float* g_sc   = (const float*)d_in[24]; const float* b_sc   = (const float*)d_in[25];
    float* out = (float*)d_out;

    // ---- workspace carve-up (bytes) ----
    char* p = (char*)d_ws;
    float* stats = (float*)p;                 p += 4096 * sizeof(float);
    float* st_mlp1 = stats + 0*512, *st_bb1 = stats + 1*512, *st_a1m = stats + 2*512;
    float* st_bb2  = stats + 3*512, *st_a2m = stats + 4*512;
    float* st_mlp2 = stats + 5*512, *st_sc  = stats + 6*512;
    _Float16* rp16    = (_Float16*)p; p += (size_t)MNK * 32 * 2;
    _Float16* fpc16   = (_Float16*)p; p += (size_t)MP  * 64 * 2;
    _Float16* fx1     = (_Float16*)p; p += (size_t)MNK * 64 * 2;
    _Float16* fx2     = (_Float16*)p; p += (size_t)MNK * 64 * 2;
    _Float16* agg16   = (_Float16*)p; p += (size_t)MP  * 128 * 2;  // agg1 & agg2
    _Float16* fagg16  = (_Float16*)p; p += (size_t)MP  * 64 * 2;
    _Float16* fpc2_16 = (_Float16*)p; p += (size_t)MP  * 128 * 2;
    float* lin_big = (float*)p;       p += (size_t)MNK * 64 * 4;   // bb1 & bb2 raw
    float* lin_pt  = (float*)p;       p += (size_t)MP  * 256 * 4;  // point-conv raw
    float* y1      = (float*)p;       p += (size_t)MP  * 256 * 4;
    float* y2      = (float*)p;       p += (size_t)MP  * 256 * 4;

    zero_stats_kernel<<<16, 256, 0, stream>>>(stats);
    relpos_kernel<<<(MNK + 255) / 256, 256, 0, stream>>>(xyz, neigh, rp16);

    // mlp1: feature (ch-major f32) -> lin_pt [MP,64] -> fpc16
    gemm_bn_kernel<64, 64, 1, 64><<<dim3(MP / 128, 1), 256, 0, stream>>>(
        nullptr, feature, NN, W_mlp1, lin_pt, st_mlp1, 64, MP);
    // bb1: rp16 -> lin_big [MNK,64] -> fx1
    gemm_bn_kernel<32, 64, 0, 10><<<dim3(MNK / 128, 1), 256, 0, stream>>>(
        rp16, nullptr, NN, W_bb1, lin_big, st_bb1, 64, MNK);
    bn_finalize_kernel<<<(MP * 64 + 255) / 256, 256, 0, stream>>>(
        lin_pt, fpc16, st_mlp1, g_mlp1, b_mlp1, 64, MP, 1);
    bn_finalize_kernel<<<(MNK * 64 + 255) / 256, 256, 0, stream>>>(
        lin_big, fx1, st_bb1, g_bb1, b_bb1, 64, MNK, 1);

    // attentive pooling 1 (fused concat+scores+softmax+sum) -> agg16 [MP,128]
    att_pool_kernel<<<MP / 8, 256, 0, stream>>>(fpc16, fx1, neigh, W_a1fc, agg16);
    // att1 mlp: agg16 -> lin_pt [MP,64] -> fagg16
    gemm_bn_kernel<128, 64, 0, 128><<<dim3(MP / 128, 1), 256, 0, stream>>>(
        agg16, nullptr, NN, W_a1m, lin_pt, st_a1m, 64, MP);
    bn_finalize_kernel<<<(MP * 64 + 255) / 256, 256, 0, stream>>>(
        lin_pt, fagg16, st_a1m, g_a1, b_a1, 64, MP, 1);

    // bb2: fx1 -> lin_big [MNK,64] -> fx2
    gemm_bn_kernel<64, 64, 0, 64><<<dim3(MNK / 128, 1), 256, 0, stream>>>(
        fx1, nullptr, NN, W_bb2, lin_big, st_bb2, 64, MNK);
    bn_finalize_kernel<<<(MNK * 64 + 255) / 256, 256, 0, stream>>>(
        lin_big, fx2, st_bb2, g_bb2, b_bb2, 64, MNK, 1);

    // attentive pooling 2 -> agg16 [MP,128] (agg1 dead)
    att_pool_kernel<<<MP / 8, 256, 0, stream>>>(fagg16, fx2, neigh, W_a2fc, agg16);
    // att2 mlp: agg16 -> lin_pt [MP,128] -> fpc2_16
    gemm_bn_kernel<128, 128, 0, 128><<<dim3(MP / 128, 1), 256, 0, stream>>>(
        agg16, nullptr, NN, W_a2m, lin_pt, st_a2m, 128, MP);
    bn_finalize_kernel<<<(MP * 128 + 255) / 256, 256, 0, stream>>>(
        lin_pt, fpc2_16, st_a2m, g_a2, b_a2, 128, MP, 1);

    // mlp2: fpc2_16 -> y1 [MP,256] (two 128-wide chunks)
    gemm_bn_kernel<128, 128, 0, 128><<<dim3(MP / 128, 2), 256, 0, stream>>>(
        fpc2_16, nullptr, NN, W_mlp2, y1, st_mlp2, 256, MP);
    // shortcut: feature (ch-major f32) -> y2 [MP,256]
    gemm_bn_kernel<64, 128, 1, 64><<<dim3(MP / 128, 2), 256, 0, stream>>>(
        nullptr, feature, NN, W_sc, y2, st_sc, 256, MP);

    final_kernel<<<(MP * 256 + 255) / 256, 256, 0, stream>>>(
        y1, y2, st_mlp2, st_sc, g_mlp2, b_mlp2, g_sc, b_sc, out);
}
